// CMG_HRNN_28698971472413
// MI455X (gfx1250) — matrix operand order repaired and checked by reading, not compiled, not run on hardware
//
#include <hip/hip_runtime.h>
#include <math.h>

typedef __bf16 bf16;
typedef bf16  v16bf __attribute__((ext_vector_type(16)));
typedef float v8f   __attribute__((ext_vector_type(8)));

static __device__ __forceinline__ bf16 tobf(float x) { return (bf16)x; }
static __device__ __forceinline__ float sigm(float x) { return 1.f / (1.f + __expf(-x)); }

// ---------------------------------------------------------------------------
// bf16 WMMA GEMM:  C[M,N] = act( A[M,K] * Bw[K,N] + bias )
// A row-major bf16 (lda = K stride), Bw row-major bf16 (ldb = padded N).
// M multiple of 32, N multiple of 256, K multiple of 64.
// Block = 256 threads = 8 waves; wave w -> mtile = w&1, 64-wide n-quad = w>>1.
// Each wave computes 16x64 (4 wmmas / k-step, one shared A fragment).
// 2-stage software pipeline with STATICALLY-NAMED register stages (no dynamic
// array indexing -> no scratch spills); loads for stage s+1 are issued before
// the wmmas of stage s.
// act: 0 = none, 1 = tanh, 2 = sigmoid
// ---------------------------------------------------------------------------
__launch_bounds__(256)
__global__ void k_gemm(const bf16* __restrict__ A, int lda,
                       const bf16* __restrict__ Bw, int ldb,
                       const float* __restrict__ bias, int nbias,
                       float* __restrict__ C, int ldc,
                       int M, int N, int K, int act)
{
    const int tid = threadIdx.x;
    const int lid = tid & 31;
    const int w   = tid >> 5;
    const int m0  = blockIdx.y * 32 + (w & 1) * 16;
    const int n0  = blockIdx.x * 256 + (w >> 1) * 64;

    const int row = m0 + (lid & 15);
    const int g   = lid >> 4;

    union U { uint4 q[2]; v16bf v; };
    v8f acc[4] = {};

    // A fragment: lane = M row (m0 + lid&15); halves = K with 8-wide group split
    const bf16* ap = A + (size_t)row * lda + g * 8;
    // B fragment: lane = N column (n0 + lid&15); halves = K with 8-wide group split
    const bf16* bp = Bw + (size_t)(g * 8) * ldb + n0 + (lid & 15);

    U a0, a1, b0[4], b1[4];
    // prologue: stage 0 (k = 0)
    a0.q[0] = *(const uint4*)(ap);
    a0.q[1] = *(const uint4*)(ap + 16);
#pragma unroll
    for (int j = 0; j < 4; ++j) {
#pragma unroll
        for (int s = 0; s < 16; ++s)
            b0[j].v[s] = bp[(size_t)(s < 8 ? s : s + 8) * ldb + j * 16];
    }

    for (int k0 = 0; k0 < K; k0 += 64) {
        // ---- load stage 1 (k0+32): always in range because K % 64 == 0 ----
        {
            const bf16* apk = ap + k0 + 32;
            const bf16* bpk = bp + (size_t)(k0 + 32) * ldb;
            a1.q[0] = *(const uint4*)(apk);
            a1.q[1] = *(const uint4*)(apk + 16);
#pragma unroll
            for (int j = 0; j < 4; ++j) {
#pragma unroll
                for (int s = 0; s < 16; ++s)
                    b1[j].v[s] = bpk[(size_t)(s < 8 ? s : s + 8) * ldb + j * 16];
            }
        }
        // ---- compute stage 0 ----
#pragma unroll
        for (int j = 0; j < 4; ++j)
            acc[j] = __builtin_amdgcn_wmma_f32_16x16x32_bf16(
                         false, a0.v, false, b0[j].v, (short)0, acc[j], false, false);
        // ---- load stage 0 for next iteration (k0+64) ----
        if (k0 + 64 < K) {
            const bf16* apk = ap + k0 + 64;
            const bf16* bpk = bp + (size_t)(k0 + 64) * ldb;
            a0.q[0] = *(const uint4*)(apk);
            a0.q[1] = *(const uint4*)(apk + 16);
#pragma unroll
            for (int j = 0; j < 4; ++j) {
#pragma unroll
                for (int s = 0; s < 16; ++s)
                    b0[j].v[s] = bpk[(size_t)(s < 8 ? s : s + 8) * ldb + j * 16];
            }
            __builtin_prefetch(bp + (size_t)(k0 + 96) * ldb, 0, 1);
        }
        // ---- compute stage 1 ----
#pragma unroll
        for (int j = 0; j < 4; ++j)
            acc[j] = __builtin_amdgcn_wmma_f32_16x16x32_bf16(
                         false, a1.v, false, b1[j].v, (short)0, acc[j], false, false);
    }

#pragma unroll
    for (int j = 0; j < 4; ++j) {
        const int nn = n0 + j * 16 + (lid & 15);
        float bv = 0.f;
        if (bias != nullptr && nn < nbias) bv = bias[nn];
#pragma unroll
        for (int r = 0; r < 8; ++r) {
            int m = m0 + r + g * 8;
            float v = acc[j][r] + bv;
            if (act == 1)      v = tanhf(v);
            else if (act == 2) v = sigm(v);
            C[(size_t)m * ldc + nn] = v;
        }
    }
}

// ------------------- weight conversion -------------------------------------
// Wt[(k+koff)*Npad + n] = bf16(W[n*K + k]); pads (n >= N) set to 0.
__global__ void k_cvtT(const float* __restrict__ W, bf16* __restrict__ Wt,
                       int N, int K, int Npad, int koff)
{
    size_t i = (size_t)blockIdx.x * blockDim.x + threadIdx.x;
    size_t tot = (size_t)K * Npad;
    if (i >= tot) return;
    int n = (int)(i % Npad);
    int k = (int)(i / Npad);
    bf16 v = (bf16)0.f;
    if (n < N) v = tobf(W[(size_t)n * K + k]);
    Wt[(size_t)(k + koff) * Npad + n] = v;
}

// plain elementwise f32 -> bf16 (weights already stored [K,N])
__global__ void k_cvtP(const float* __restrict__ W, bf16* __restrict__ o, size_t n)
{
    size_t i = (size_t)blockIdx.x * blockDim.x + threadIdx.x;
    if (i < n) o[i] = tobf(W[i]);
}

__global__ void k_zero_f(float* __restrict__ p, size_t n)
{
    size_t i = (size_t)blockIdx.x * blockDim.x + threadIdx.x;
    if (i < n) p[i] = 0.f;
}

// ------------------- per-event prep ----------------------------------------
__global__ void k_prep_event(const float* __restrict__ event,
                             const int* __restrict__ seq,
                             const int* __restrict__ esi, int e,
                             float* __restrict__ last,      // read, then reset to 0 (lss)
                             const float* __restrict__ sh,
                             float* __restrict__ ev_f, bf16* __restrict__ ev_bf,
                             bf16* __restrict__ last_bf, bf16* __restrict__ sh_bf,
                             int* __restrict__ slen)
{
    int i = blockIdx.x * blockDim.x + threadIdx.x;
    if (i >= 32 * 1024) return;
    int b = i >> 10, k = i & 1023;
    int idx = esi[b * 8 + e];
    float evv = event[(size_t)idx * 1024 + k];
    ev_f[i]    = evv;
    ev_bf[i]   = tobf(evv);
    last_bf[i] = tobf(last[i]);
    sh_bf[i]   = tobf(sh[i]);
    last[i]    = 0.f;                       // lss starts at zero each event
    if (k == 0) {
        int c = 0;
        for (int j = 0; j < 21; ++j) c += (seq[(size_t)idx * 21 + j] > 0) ? 1 : 0;
        slen[b] = c + 2;
    }
}

__global__ void k_gather_clip(const float* __restrict__ clip,
                              const int* __restrict__ esi, int e,
                              bf16* __restrict__ cl_bf)
{
    size_t i = (size_t)blockIdx.x * blockDim.x + threadIdx.x;
    if (i >= (size_t)32 * 50 * 1024) return;
    int b = (int)(i / (50 * 1024));
    size_t rest = i % (50 * 1024);
    int idx = esi[b * 8 + e];
    cl_bf[i] = tobf(clip[(size_t)idx * 50 * 1024 + rest]);
}

__global__ void k_build_cat3(const bf16* __restrict__ sh_bf,
                             const float* __restrict__ pp,
                             const float* __restrict__ ep,
                             bf16* __restrict__ catA)
{
    int i = blockIdx.x * blockDim.x + threadIdx.x;
    if (i >= 32 * 3072) return;
    int b = i / 3072, k = i % 3072;
    bf16 v;
    if (k < 1024)      v = sh_bf[b * 1024 + k];
    else if (k < 2048) v = tobf(pp[b * 1024 + k - 1024]);
    else               v = tobf(ep[b * 1024 + k - 2048]);
    catA[i] = v;
}

__global__ void k_build_xs(const float* __restrict__ pp,
                           const float* __restrict__ ep,
                           const float* __restrict__ gf,
                           const bf16* __restrict__ sh_bf,
                           bf16* __restrict__ xs)
{
    int i = blockIdx.x * blockDim.x + threadIdx.x;
    if (i >= 32 * 3072) return;
    int b = i / 3072, k = i % 3072;
    bf16 v;
    if (k < 1024)      v = tobf(pp[b * 1024 + k] * gf[b * 1024 + k]);
    else if (k < 2048) v = tobf(ep[b * 1024 + k - 1024] * (1.f - gf[b * 1024 + k - 1024]));
    else               v = sh_bf[b * 1024 + k - 2048];
    xs[i] = v;
}

// LSTM cell (pytorch gate order i,f,g,o). Optional bf16 mirrors + lss update.
__global__ void k_lstm_cell(const float* __restrict__ gates,
                            float* __restrict__ h, float* __restrict__ c,
                            bf16* __restrict__ h_bf, bf16* __restrict__ c_bf,
                            float* __restrict__ lss, const int* __restrict__ slen, int t)
{
    int i = blockIdx.x * blockDim.x + threadIdx.x;
    if (i >= 32 * 1024) return;
    int b = i >> 10, r = i & 1023;
    const float* gb = gates + (size_t)b * 4096;
    float ig = sigm(gb[r]);
    float fg = sigm(gb[r + 1024]);
    float gg = tanhf(gb[r + 2048]);
    float og = sigm(gb[r + 3072]);
    float c2 = fg * c[i] + ig * gg;
    float h2 = og * tanhf(c2);
    c[i] = c2; h[i] = h2;
    if (h_bf) h_bf[i] = tobf(h2);
    if (c_bf) c_bf[i] = tobf(c2);
    if (lss && slen[b] == t + 2) lss[i] = h2;
}

__global__ void k_init_hc(const float* __restrict__ hin,
                          float* __restrict__ h, float* __restrict__ c,
                          bf16* __restrict__ h_bf)
{
    int i = blockIdx.x * blockDim.x + threadIdx.x;
    if (i >= 32 * 1024) return;
    float v = hin[i];
    h[i] = v; c[i] = v; h_bf[i] = tobf(v);
}

// Attention + embed + build core-LSTM input [xt | att_feat | ev | h] (bf16)
__launch_bounds__(256)
__global__ void k_attention(const float* __restrict__ att_pre,   // [32,50,512]
                            const float* __restrict__ hWh,       // [32,512]
                            const float* __restrict__ att_v,     // [512]
                            const float* __restrict__ clip,      // [64,50,1024]
                            const float* __restrict__ clip_mask, // [64,50]
                            const int* __restrict__ esi, int e, int t,
                            const int* __restrict__ seq,         // [64,21]
                            const float* __restrict__ embed_W,   // [10001,512]
                            const float* __restrict__ ev_f,      // [32,1024]
                            const float* __restrict__ h_f,       // [32,1024]
                            bf16* __restrict__ xcore)            // [32,3584]
{
    __shared__ float s_att[50];
    int b = blockIdx.x;
    int tid = threadIdx.x, lid = tid & 31, w = tid >> 5;
    int idx = esi[b * 8 + e];
    const float* ap = att_pre + (size_t)b * 50 * 512;
    const float* hw = hWh + (size_t)b * 512;

    for (int j = w; j < 50; j += 8) {
        float s = 0.f;
        const float* apj = ap + (size_t)j * 512;
        for (int a = lid; a < 512; a += 32)
            s += tanhf(apj[a] + hw[a]) * att_v[a];
        for (int o = 16; o > 0; o >>= 1) s += __shfl_down(s, o, 32);
        if (lid == 0) {
            float m = clip_mask[(size_t)idx * 50 + j];
            s_att[j] = (m > 0.f) ? s : -1e9f;
        }
    }
    __syncthreads();
    if (w == 0) {
        float mx = -INFINITY;
        for (int j = lid; j < 50; j += 32) mx = fmaxf(mx, s_att[j]);
        for (int o = 16; o > 0; o >>= 1) mx = fmaxf(mx, __shfl_down(mx, o, 32));
        mx = __shfl(mx, 0, 32);
        float sm = 0.f;
        float ex[2];
        for (int j = lid, q = 0; j < 50; j += 32, ++q) { ex[q] = __expf(s_att[j] - mx); sm += ex[q]; }
        for (int o = 16; o > 0; o >>= 1) sm += __shfl_down(sm, o, 32);
        sm = __shfl(sm, 0, 32);
        float inv = 1.f / sm;
        for (int j = lid, q = 0; j < 50; j += 32, ++q) s_att[j] = ex[q] * inv;
    }
    __syncthreads();

    const float* clb = clip + (size_t)idx * 50 * 1024;
    bf16* xc = xcore + (size_t)b * 3584;
    for (int d = tid; d < 1024; d += 256) {
        float s = 0.f;
        for (int j = 0; j < 50; ++j) s += s_att[j] * clb[(size_t)j * 1024 + d];
        xc[512 + d]  = tobf(s);
        xc[1536 + d] = tobf(ev_f[b * 1024 + d]);
        xc[2560 + d] = tobf(h_f[b * 1024 + d]);
    }
    int it = seq[(size_t)idx * 21 + t];
    const float* em = embed_W + (size_t)it * 512;
    for (int d = tid; d < 512; d += 256) xc[d] = tobf(em[d]);
}

__launch_bounds__(256)
__global__ void k_logsoftmax(const float* __restrict__ logits, // [32,10240]
                             float* __restrict__ out, int e, int t)
{
    __shared__ float red[8];
    int b = blockIdx.x, tid = threadIdx.x, lid = tid & 31, w = tid >> 5;
    const float* lg = logits + (size_t)b * 10240;

    float mx = -INFINITY;
    for (int v = tid; v < 10001; v += 256) mx = fmaxf(mx, lg[v]);
    for (int o = 16; o > 0; o >>= 1) mx = fmaxf(mx, __shfl_down(mx, o, 32));
    if (lid == 0) red[w] = mx;
    __syncthreads();
    mx = red[0];
    for (int i = 1; i < 8; ++i) mx = fmaxf(mx, red[i]);
    __syncthreads();

    float sm = 0.f;
    for (int v = tid; v < 10001; v += 256) sm += __expf(lg[v] - mx);
    for (int o = 16; o > 0; o >>= 1) sm += __shfl_down(sm, o, 32);
    if (lid == 0) red[w] = sm;
    __syncthreads();
    float tot = 0.f;
    for (int i = 0; i < 8; ++i) tot += red[i];
    float lse = mx + __logf(tot);

    float* ob = out + (((size_t)b * 8 + e) * 20 + t) * 10001;
    for (int v = tid; v < 10001; v += 256) ob[v] = lg[v] - lse;
}

// ---------------------------------------------------------------------------
extern "C" void kernel_launch(void* const* d_in, const int* in_sizes, int n_in,
                              void* d_out, int out_size, void* d_ws, size_t ws_size,
                              hipStream_t stream)
{
    (void)in_sizes; (void)n_in; (void)out_size; (void)ws_size;
    const float* event     = (const float*)d_in[0];
    const float* clip      = (const float*)d_in[1];
    const float* clip_mask = (const float*)d_in[2];
    const int*   seq       = (const int*)  d_in[3];
    const int*   esi       = (const int*)  d_in[4];
    const float* embed_W   = (const float*)d_in[5];
    const float* gproj_w   = (const float*)d_in[6];
    const float* gproj_b   = (const float*)d_in[7];
    const float* lproj_w   = (const float*)d_in[8];
    const float* lproj_b   = (const float*)d_in[9];
    const float* gate_w    = (const float*)d_in[10];
    const float* gate_b    = (const float*)d_in[11];
    const float* sent_Wih  = (const float*)d_in[12];
    const float* sent_Whh  = (const float*)d_in[13];
    const float* trans_w   = (const float*)d_in[14];
    const float* trans_b   = (const float*)d_in[15];
    const float* att_Wa    = (const float*)d_in[16];
    const float* att_Wh    = (const float*)d_in[17];
    const float* att_v     = (const float*)d_in[18];
    const float* core_Wih  = (const float*)d_in[19];
    const float* core_Whh  = (const float*)d_in[20];
    const float* core_b    = (const float*)d_in[21];
    const float* logit_w   = (const float*)d_in[22];
    const float* logit_b   = (const float*)d_in[23];
    float* out = (float*)d_out;

    const int NLOG = 10240;   // 10001 padded to multiple of 256

    char* p = (char*)d_ws;
    auto alloc = [&](size_t bytes) -> char* {
        char* r = p; p += (bytes + 255) & ~(size_t)255; return r;
    };
    // bf16 transposed / packed weights
    bf16* wt_gproj = (bf16*)alloc((size_t)1024 * 1024 * 2);
    bf16* wt_lproj = (bf16*)alloc((size_t)1024 * 1024 * 2);
    bf16* wt_gate  = (bf16*)alloc((size_t)3072 * 1024 * 2);
    bf16* wt_sent  = (bf16*)alloc((size_t)3072 * 4096 * 2);
    bf16* wt_trans = (bf16*)alloc((size_t)1024 * 1024 * 2);
    bf16* wt_attWa = (bf16*)alloc((size_t)1024 * 512  * 2);
    bf16* wt_attWh = (bf16*)alloc((size_t)1024 * 512  * 2);
    bf16* wt_core  = (bf16*)alloc((size_t)3584 * 4096 * 2);
    bf16* wt_logit = (bf16*)alloc((size_t)1024 * NLOG * 2);
    // carries
    float* carry = (float*)alloc((size_t)3 * 32 * 1024 * 4);
    float* sh = carry, *sc = carry + 32 * 1024, *last = carry + 2 * 32 * 1024;
    // activations
    bf16*  sh_bf   = (bf16*) alloc((size_t)32 * 1024 * 2);
    bf16*  last_bf = (bf16*) alloc((size_t)32 * 1024 * 2);
    bf16*  sc_bf   = (bf16*) alloc((size_t)32 * 1024 * 2);
    bf16*  ev_bf   = (bf16*) alloc((size_t)32 * 1024 * 2);
    bf16*  h_bf    = (bf16*) alloc((size_t)32 * 1024 * 2);
    float* ev_f    = (float*)alloc((size_t)32 * 1024 * 4);
    bf16*  cl_bf   = (bf16*) alloc((size_t)32 * 50 * 1024 * 2);
    float* pp_f    = (float*)alloc((size_t)32 * 1024 * 4);
    float* ep_f    = (float*)alloc((size_t)32 * 1024 * 4);
    float* g_f     = (float*)alloc((size_t)32 * 1024 * 4);
    bf16*  catA    = (bf16*) alloc((size_t)32 * 3072 * 2);
    bf16*  xs_bf   = (bf16*) alloc((size_t)32 * 3072 * 2);
    float* gates_f = (float*)alloc((size_t)32 * 4096 * 4);
    float* h_f     = (float*)alloc((size_t)32 * 1024 * 4);
    float* c_f     = (float*)alloc((size_t)32 * 1024 * 4);
    float* att_pre = (float*)alloc((size_t)32 * 50 * 512 * 4);
    float* hWh     = (float*)alloc((size_t)32 * 512 * 4);
    bf16*  xcore   = (bf16*) alloc((size_t)32 * 3584 * 2);
    float* logits  = (float*)alloc((size_t)32 * NLOG * 4);
    int*   slen    = (int*)  alloc((size_t)32 * 4);

    auto blk = [](size_t n) { return (unsigned)((n + 255) / 256); };
    auto cvtT = [&](const float* W, bf16* Wt, int N, int K, int Npad, int koff) {
        k_cvtT<<<blk((size_t)K * Npad), 256, 0, stream>>>(W, Wt, N, K, Npad, koff);
    };
    auto gemm = [&](const bf16* A, int lda, const bf16* Bw, int ldb,
                    const float* bias, int nbias, float* C, int ldc,
                    int M, int N, int K, int act) {
        dim3 g((unsigned)(N / 256), (unsigned)((M + 31) / 32));
        k_gemm<<<g, 256, 0, stream>>>(A, lda, Bw, ldb, bias, nbias, C, ldc, M, N, K, act);
    };

    // ---- one-time (per launch) weight conversion --------------------------
    cvtT(gproj_w,  wt_gproj, 1024, 1024, 1024, 0);
    cvtT(lproj_w,  wt_lproj, 1024, 1024, 1024, 0);
    cvtT(gate_w,   wt_gate,  1024, 3072, 1024, 0);
    cvtT(sent_Wih, wt_sent,  4096, 2048, 4096, 0);
    cvtT(sent_Whh, wt_sent,  4096, 1024, 4096, 2048);
    cvtT(trans_w,  wt_trans, 1024, 1024, 1024, 0);
    k_cvtP<<<blk((size_t)1024 * 512), 256, 0, stream>>>(att_Wa, wt_attWa, (size_t)1024 * 512);
    k_cvtP<<<blk((size_t)1024 * 512), 256, 0, stream>>>(att_Wh, wt_attWh, (size_t)1024 * 512);
    cvtT(core_Wih, wt_core, 4096, 2560, 4096, 0);
    cvtT(core_Whh, wt_core, 4096, 1024, 4096, 2560);
    cvtT(logit_w,  wt_logit, 10001, 1024, NLOG, 0);

    k_zero_f<<<blk((size_t)3 * 32 * 1024), 256, 0, stream>>>(carry, (size_t)3 * 32 * 1024);

    // ---- event loop -------------------------------------------------------
    for (int e = 0; e < 8; ++e) {
        k_prep_event<<<blk(32 * 1024), 256, 0, stream>>>(
            event, seq, esi, e, last, sh, ev_f, ev_bf, last_bf, sh_bf, slen);
        k_gather_clip<<<blk((size_t)32 * 50 * 1024), 256, 0, stream>>>(clip, esi, e, cl_bf);

        gemm(last_bf, 1024, wt_gproj, 1024, gproj_b, 1024, pp_f, 1024, 32, 1024, 1024, 1);
        gemm(ev_bf,   1024, wt_lproj, 1024, lproj_b, 1024, ep_f, 1024, 32, 1024, 1024, 1);
        k_build_cat3<<<blk(32 * 3072), 256, 0, stream>>>(sh_bf, pp_f, ep_f, catA);
        gemm(catA, 3072, wt_gate, 1024, gate_b, 1024, g_f, 1024, 32, 1024, 3072, 2);
        k_build_xs<<<blk(32 * 3072), 256, 0, stream>>>(pp_f, ep_f, g_f, sh_bf, xs_bf);
        gemm(xs_bf, 3072, wt_sent, 4096, nullptr, 0, gates_f, 4096, 32, 4096, 3072, 0);
        k_lstm_cell<<<blk(32 * 1024), 256, 0, stream>>>(
            gates_f, sh, sc, (bf16*)nullptr, sc_bf, (float*)nullptr, (const int*)nullptr, 0);
        gemm(sc_bf, 1024, wt_trans, 1024, trans_b, 1024, h_f, 1024, 32, 1024, 1024, 0);
        k_init_hc<<<blk(32 * 1024), 256, 0, stream>>>(h_f, h_f, c_f, h_bf);
        gemm(cl_bf, 1024, wt_attWa, 512, nullptr, 0, att_pre, 512, 1600, 512, 1024, 0);

        for (int t = 0; t < 20; ++t) {
            gemm(h_bf, 1024, wt_attWh, 512, nullptr, 0, hWh, 512, 32, 512, 1024, 0);
            k_attention<<<32, 256, 0, stream>>>(att_pre, hWh, att_v, clip, clip_mask,
                                                esi, e, t, seq, embed_W, ev_f, h_f, xcore);
            gemm(xcore, 3584, wt_core, 4096, core_b, 4096, gates_f, 4096, 32, 4096, 3584, 0);
            k_lstm_cell<<<blk(32 * 1024), 256, 0, stream>>>(
                gates_f, h_f, c_f, h_bf, (bf16*)nullptr, last, slen, t);
            gemm(h_bf, 1024, wt_logit, NLOG, logit_b, 10001, logits, NLOG, 32, NLOG, 1024, 0);
            k_logsoftmax<<<32, 256, 0, stream>>>(logits, out, e, t);
        }
    }
}
